// KnnXLMultiHeadsAttention_53085795779503
// MI455X (gfx1250) — matrix operand order repaired and checked by reading, not compiled, not run on hardware
//
#include <hip/hip_runtime.h>
#include <hip/hip_bf16.h>

typedef __attribute__((ext_vector_type(16))) _Float16 v16h;
typedef __attribute__((ext_vector_type(8)))  float    v8f;

static __device__ __forceinline__ v8f wmma_f16(const v16h& a, const v16h& b, const v8f& c) {
    return __builtin_amdgcn_wmma_f32_16x16x32_f16(false, a, false, b, (short)0, c, false, false);
}

// ---------------------------------------------------------------------------
// Kernel 1: fused QKV projection.  C[m,n] = x[m,:] . W[n,:] + bias[n]
// One wave per 16x16 output tile; 16 WMMA K-steps of 32.
// q -> scratch [b,h,s,d];  k,v -> d_out kv region [b,s,2,512] (= current_kv_memory)
// ---------------------------------------------------------------------------
__global__ __launch_bounds__(128) void qkv_gemm_kernel(
    const float* __restrict__ x,
    const float* __restrict__ Wq, const float* __restrict__ bq,
    const float* __restrict__ Wk, const float* __restrict__ bk,
    const float* __restrict__ Wv, const float* __restrict__ bv,
    float* __restrict__ qbuf, float* __restrict__ kvout)
{
    const int lane = threadIdx.x & 31;
    const int r  = lane & 15;
    const int hi = lane >> 4;
    const int gw = blockIdx.x * 4 + (threadIdx.x >> 5);   // 24576 tiles total
    const int nt = gw % 96;                               // 96 col tiles (3 weights x 32)
    const int mt = gw / 96;                               // 256 row tiles
    const int sel = nt >> 5;                              // 0=q 1=k 2=v
    const int nct = nt & 31;

    const float* W    = (sel == 0) ? Wq : ((sel == 1) ? Wk : Wv);
    const float* bias = (sel == 0) ? bq : ((sel == 1) ? bk : bv);

    const int nc = nct * 16 + r;                 // output column owned by this lane (B operand)
    const float* wrow = W + (size_t)nc * 512;
    const float* xrow = x + (size_t)(mt * 16 + r) * 512;

    v8f c = {0.f, 0.f, 0.f, 0.f, 0.f, 0.f, 0.f, 0.f};
    for (int ek = 0; ek < 16; ++ek) {
        const int e0 = ek * 32;
        v16h a, b;
#pragma unroll
        for (int e = 0; e < 16; ++e) {
            const int ka = ((e & 8) << 1) + hi * 8 + (e & 7);   // A layout K index
            a[e] = (_Float16)xrow[e0 + ka];
            b[e] = (_Float16)wrow[e0 + ka];            // B layout K index
        }
        c = wmma_f16(a, b, c);
    }

    const float bb = bias[nc];
#pragma unroll
    for (int v = 0; v < 8; ++v) {
        const int m = mt * 16 + v + hi * 8;       // global row = b*1024 + s
        const int bidx = m >> 10, s = m & 1023;
        const float val = c[v] + bb;
        if (sel == 0) {
            const int hh = nc >> 6, d = nc & 63;
            qbuf[(((size_t)bidx * 8 + hh) * 1024 + s) * 64 + d] = val;
        } else {
            kvout[(((size_t)bidx * 1024 + s) * 2 + (sel - 1)) * 512 + nc] = val;
        }
    }
}

// ---------------------------------------------------------------------------
// Kernel 2: causal XL flash attention.  One wave per 16-query tile of one (b,h).
// Computes S^T = K.Q^T (C frag: row=key, col=query -> lane owns one query),
// online softmax per lane, P^T built via shfl_xor(16), O^T = V^T . P^T.
// ---------------------------------------------------------------------------
__global__ __launch_bounds__(128) void xl_attn_kernel(
    const float* __restrict__ qbuf,   // [b,h,s,64]
    const float* __restrict__ xlmem,  // [b,1024,2,512]
    const float* __restrict__ curkv,  // [b,1024,2,512] (written by kernel 1 into d_out)
    const float* __restrict__ rel,    // [8,1024,2048]
    float* __restrict__ att)          // [b,h,s,64]
{
    const int lane = threadIdx.x & 31;
    const int r  = lane & 15;
    const int hi = lane >> 4;
    const int gw = blockIdx.x * 4 + (threadIdx.x >> 5);   // 2048 waves
    const int qt = gw & 63;
    const int h  = (gw >> 6) & 7;
    const int b  = gw >> 9;
    const int i0 = qt * 16;
    const float scale = 0.125f;                            // 64^-0.5
    const int qi = i0 + r;                                 // query owned by this lane

    // Q^T B-frags: lane = query column, features ((e&8)<<1)+hi*8+(e&7) per 32-chunk
    const float* qrow = qbuf + (((size_t)b * 8 + h) * 1024 + qi) * 64;
    v16h qb0, qb1;
#pragma unroll
    for (int e = 0; e < 16; ++e) {
        qb0[e] = (_Float16)qrow[((e & 8) << 1) + hi * 8 + (e & 7)];
        qb1[e] = (_Float16)qrow[32 + ((e & 8) << 1) + hi * 8 + (e & 7)];
    }

    float m_run = -3.0e38f, l_run = 0.f;
    v8f o0 = {0.f,0.f,0.f,0.f,0.f,0.f,0.f,0.f};
    v8f o1 = o0, o2 = o0, o3 = o0;

    const int nkeys  = 1024 + i0 + 16;          // max key count for this tile (causal)
    const int ntiles = (nkeys + 31) >> 5;

    for (int t = 0; t < ntiles; ++t) {
        const int j0 = t * 32;
        const float* base = (j0 < 1024)
            ? (xlmem + ((size_t)(b * 1024 + j0) * 2) * 512 + h * 64)
            : (curkv + ((size_t)(b * 1024 + (j0 - 1024)) * 2) * 512 + h * 64);
        // key row stride = 1024 floats; values at +512

        // K-tile A-frags (lane = key row)
        const float* krow_a = base + (size_t)r * 1024;
        const float* krow_b = base + (size_t)(r + 16) * 1024;
        v16h ka0, ka1, kb0, kb1;
#pragma unroll
        for (int e = 0; e < 16; ++e) {
            const int kf = ((e & 8) << 1) + hi * 8 + (e & 7);
            ka0[e] = (_Float16)krow_a[kf];
            ka1[e] = (_Float16)krow_a[32 + kf];
            kb0[e] = (_Float16)krow_b[kf];
            kb1[e] = (_Float16)krow_b[32 + kf];
        }
        const v8f zero = {0.f,0.f,0.f,0.f,0.f,0.f,0.f,0.f};
        v8f ct0 = wmma_f16(ka0, qb0, zero); ct0 = wmma_f16(ka1, qb1, ct0);
        v8f ct1 = wmma_f16(kb0, qb0, zero); ct1 = wmma_f16(kb1, qb1, ct1);

        // logits: ct0[v] -> (key j0+v+8*hi, query qi); rel added before scale, then mask
        const float* relrow = rel + ((size_t)h * 1024 + qi) * 2048 + j0;
        float s0[8], s1[8];
        float tmax = -3.0e38f;
#pragma unroll
        for (int v = 0; v < 8; ++v) {
            const int k0 = v + hi * 8;
            const int k1 = k0 + 16;
            float a0 = (ct0[v] + relrow[k0]) * scale;
            float a1 = (ct1[v] + relrow[k1]) * scale;
            if (j0 + k0 > 1024 + qi) a0 = -3.0e38f;
            if (j0 + k1 > 1024 + qi) a1 = -3.0e38f;
            s0[v] = a0; s1[v] = a1;
            tmax = fmaxf(tmax, fmaxf(a0, a1));
        }
        tmax = fmaxf(tmax, __shfl_xor(tmax, 16, 32));   // combine half-waves (same query)

        const float m_new = fmaxf(m_run, tmax);
        const float alpha = __expf(m_run - m_new);
        float p0[8], p1[8], lsum = 0.f;
#pragma unroll
        for (int v = 0; v < 8; ++v) {
            p0[v] = __expf(s0[v] - m_new);
            p1[v] = __expf(s1[v] - m_new);
            lsum += p0[v] + p1[v];
        }
        lsum += __shfl_xor(lsum, 16, 32);
        l_run = l_run * alpha + lsum;
        m_run = m_new;

#pragma unroll
        for (int v = 0; v < 8; ++v) { o0[v] *= alpha; o1[v] *= alpha; o2[v] *= alpha; o3[v] *= alpha; }

        // Build P^T B-frag (32 keys x 16 queries): lane = query, K = key (hi*16+e)
        v16h pb;
#pragma unroll
        for (int v = 0; v < 8; ++v) {
            const float x0 = __shfl_xor(p0[v], 16, 32);
            const float x1 = __shfl_xor(p1[v], 16, 32);
            pb[v]     = (_Float16)(hi ? x1    : p0[v]);   // keys hi*16 + v
            pb[v + 8] = (_Float16)(hi ? p1[v] : x0);      // keys hi*16 + 8 + v
        }

        // O^T += V^T . P^T, per 16-dim chunk
        const float* vbase = base + 512;
#pragma unroll
        for (int cix = 0; cix < 4; ++cix) {
            v16h va;
            const int dim = cix * 16 + r;
#pragma unroll
            for (int e = 0; e < 16; ++e) {
                const int key = hi * 16 + e;
                va[e] = (_Float16)vbase[(size_t)key * 1024 + dim];
            }
            if      (cix == 0) o0 = wmma_f16(va, pb, o0);
            else if (cix == 1) o1 = wmma_f16(va, pb, o1);
            else if (cix == 2) o2 = wmma_f16(va, pb, o2);
            else               o3 = wmma_f16(va, pb, o3);
        }
    }

    const float inv = 1.f / l_run;
    float* orow = att + (((size_t)b * 8 + h) * 1024 + qi) * 64;
#pragma unroll
    for (int v = 0; v < 8; ++v) {
        orow[ 0 + v + 8 * hi] = o0[v] * inv;
        orow[16 + v + 8 * hi] = o1[v] * inv;
        orow[32 + v + 8 * hi] = o2[v] * inv;
        orow[48 + v + 8 * hi] = o3[v] * inv;
    }
}

// ---------------------------------------------------------------------------
// Kernel 3: kNN external attention + sigmoid-gated merge. One block per (b,s).
// Memory-bound (512 MB stream of knn_retrieved) -> plain coalesced VALU.
// Writes merged activations in [b*s, 512] layout ready for the output GEMM.
// ---------------------------------------------------------------------------
__global__ __launch_bounds__(256) void knn_merge_kernel(
    const float* __restrict__ qbuf, const float* __restrict__ att,
    const float* __restrict__ knn,  const float* __restrict__ gate,
    float* __restrict__ merged)
{
    __shared__ float lw[256];
    const int b = blockIdx.x >> 10, s = blockIdx.x & 1023;
    const int t = threadIdx.x;

    {   // logits[h][kk]
        const int h = t >> 5, kk = t & 31;
        const float* qr = qbuf + (((size_t)b * 8 + h) * 1024 + s) * 64;
        const float* kr = knn + (((size_t)(b * 1024 + s) * 32 + kk) * 2 + 0) * 512 + h * 64;
        float dot = 0.f;
#pragma unroll
        for (int d = 0; d < 64; d += 4)
            dot += qr[d] * kr[d] + qr[d+1] * kr[d+1] + qr[d+2] * kr[d+2] + qr[d+3] * kr[d+3];
        lw[t] = dot * 0.125f;
    }
    __syncthreads();

    if (t < 8) {   // per-head softmax over 32 logits
        float m = -3.0e38f;
#pragma unroll
        for (int k = 0; k < 32; ++k) m = fmaxf(m, lw[t * 32 + k]);
        float e[32], sum = 0.f;
#pragma unroll
        for (int k = 0; k < 32; ++k) { e[k] = __expf(lw[t * 32 + k] - m); sum += e[k]; }
        const float inv = 1.f / sum;
#pragma unroll
        for (int k = 0; k < 32; ++k) lw[t * 32 + k] = e[k] * inv;
    }
    __syncthreads();

#pragma unroll
    for (int rep = 0; rep < 2; ++rep) {
        const int idx = t + rep * 256;            // 0..511 = h*64+d
        const int h = idx >> 6, d = idx & 63;
        const float* vr = knn + (((size_t)(b * 1024 + s) * 32) * 2 + 1) * 512 + h * 64 + d;
        float acc = 0.f;
#pragma unroll
        for (int k = 0; k < 32; ++k) acc += lw[h * 32 + k] * vr[(size_t)k * 1024];
        const float g = 1.f / (1.f + __expf(-gate[h]));
        const float loc = att[(((size_t)b * 8 + h) * 1024 + s) * 64 + d];
        merged[((size_t)(b * 1024 + s)) * 512 + idx] = loc * g + acc * (1.f - g);
    }
}

// ---------------------------------------------------------------------------
// Kernel 4: output projection  out = merged @ Wo^T + bo  (WMMA f16)
// ---------------------------------------------------------------------------
__global__ __launch_bounds__(128) void out_gemm_kernel(
    const float* __restrict__ merged, const float* __restrict__ Wo,
    const float* __restrict__ bo, float* __restrict__ out)
{
    const int lane = threadIdx.x & 31;
    const int r  = lane & 15;
    const int hi = lane >> 4;
    const int gw = blockIdx.x * 4 + (threadIdx.x >> 5);   // 8192 tiles
    const int nt = gw & 31;
    const int mt = gw >> 5;

    const int nc = nt * 16 + r;
    const float* wrow = Wo + (size_t)nc * 512;
    const float* arow = merged + (size_t)(mt * 16 + r) * 512;

    v8f c = {0.f,0.f,0.f,0.f,0.f,0.f,0.f,0.f};
    for (int ek = 0; ek < 16; ++ek) {
        const int e0 = ek * 32;
        v16h a, b;
#pragma unroll
        for (int e = 0; e < 16; ++e) {
            const int ka = ((e & 8) << 1) + hi * 8 + (e & 7);
            a[e] = (_Float16)arow[e0 + ka];
            b[e] = (_Float16)wrow[e0 + ka];
        }
        c = wmma_f16(a, b, c);
    }
    const float bb = bo[nc];
#pragma unroll
    for (int v = 0; v < 8; ++v) {
        const int m = mt * 16 + v + hi * 8;
        out[(size_t)m * 512 + nc] = c[v] + bb;
    }
}

// ---------------------------------------------------------------------------
extern "C" void kernel_launch(void* const* d_in, const int* in_sizes, int n_in,
                              void* d_out, int out_size, void* d_ws, size_t ws_size,
                              hipStream_t stream) {
    const float* x     = (const float*)d_in[0];
    const float* rel   = (const float*)d_in[1];
    const float* xlmem = (const float*)d_in[2];
    const float* knn   = (const float*)d_in[3];
    const float* Wq    = (const float*)d_in[4];
    const float* bq    = (const float*)d_in[5];
    const float* Wk    = (const float*)d_in[6];
    const float* bk    = (const float*)d_in[7];
    const float* Wv    = (const float*)d_in[8];
    const float* bv    = (const float*)d_in[9];
    const float* Wo    = (const float*)d_in[10];
    const float* bo    = (const float*)d_in[11];
    const float* gate  = (const float*)d_in[12];

    float* out   = (float*)d_out;                 // [4,1024,512]
    float* kvout = out + (size_t)4 * 1024 * 512;  // [4,1024,2,512] = current_kv_memory

    float* ws     = (float*)d_ws;
    float* qbuf   = ws;                           // 2,097,152 floats  [b,h,s,d]
    float* att    = ws + (size_t)2097152;         // 2,097,152 floats  [b,h,s,d]
    float* merged = ws + (size_t)2 * 2097152;     // 2,097,152 floats  [b*s,512]

    qkv_gemm_kernel <<<6144, 128, 0, stream>>>(x, Wq, bq, Wk, bk, Wv, bv, qbuf, kvout);
    xl_attn_kernel  <<<512,  128, 0, stream>>>(qbuf, xlmem, kvout, rel, att);
    knn_merge_kernel<<<4096, 256, 0, stream>>>(qbuf, att, knn, gate, merged);
    out_gemm_kernel <<<2048, 128, 0, stream>>>(merged, Wo, bo, out);
}
